// EpipolarCrossAttention_48799418417191
// MI455X (gfx1250) — compile-verified, matrix operand order checked by reading, not run on hardware
//
#include <hip/hip_runtime.h>
#include <math.h>

#define NQ 8192
#define NM 8192
#define NC 256
#define NSPLIT 4
#define KSPAN (NM / NSPLIT)      // 2048 keys per split
#define NEGV  (-1.0e9f)
#define VTOL  3.0f
#define MAXD  192.0f
#define SCALE 0.0625f            // 1/sqrt(256), folded into Q projection

typedef __attribute__((ext_vector_type(16))) __bf16 v16bf;
typedef __attribute__((ext_vector_type(8)))  float  v8f;

union FragU { uint4 u[2]; v16bf v; };

__device__ __forceinline__ unsigned short f2bf(float f) {
    unsigned int u = __float_as_uint(f);
    u = (u + 0x7FFFu + ((u >> 16) & 1u)) >> 16;
    return (unsigned short)u;
}
__device__ __forceinline__ float bf2f(unsigned short s) {
    return __uint_as_float(((unsigned int)s) << 16);
}

// 16x32 bf16 A-layout (mirrored for B) fragment from a row-major matrix.
// lane (l = lane&15, h = lane>>4) holds row l, K runs [8h,8h+8) and
// [16+8h,16+8h+8): two 16-byte loads per lane. Works for global or LDS ptrs.
__device__ __forceinline__ v16bf load_frag(const unsigned short* base, int stride, int lane) {
    const int l = lane & 15, h = (lane >> 4) & 1;
    const unsigned short* p = base + (size_t)l * stride;
    FragU f;
    f.u[0] = *(const uint4*)(p + 8 * h);
    f.u[1] = *(const uint4*)(p + 16 + 8 * h);
    return f.v;
}

__device__ __forceinline__ float red_max16(float v) {
    #pragma unroll
    for (int o = 1; o < 16; o <<= 1) v = fmaxf(v, __shfl_xor(v, o, 32));
    return v;
}
__device__ __forceinline__ float red_sum16(float v) {
    #pragma unroll
    for (int o = 1; o < 16; o <<= 1) v += __shfl_xor(v, o, 32);
    return v;
}

// CDNA5 async global->LDS copy (GVS mode), tracked by ASYNCcnt.
// lds_addr: per-lane wave-relative LDS byte address (low 32 bits of a
// __shared__ generic pointer); sbase: uniform 64-bit base; byte_off: per-lane
// 32-bit byte offset.
__device__ __forceinline__ void async_ld_b128(unsigned int lds_addr,
                                              const unsigned short* sbase,
                                              unsigned int byte_off) {
    asm volatile("global_load_async_to_lds_b128 %0, %1, %2"
                 :
                 : "v"(lds_addr), "v"(byte_off), "s"(sbase)
                 : "memory");
}
__device__ __forceinline__ void wait_async0() {
    asm volatile("s_wait_asynccnt 0" ::: "memory");
}

// ---------------------------------------------------------------- conversions
__global__ void cvt_bf16_kernel(const float* __restrict__ in,
                                unsigned short* __restrict__ out, int n) {
    int i = blockIdx.x * blockDim.x + threadIdx.x;
    if (i < n) out[i] = f2bf(in[i]);
}

// ---------------------------------------------------- projection  Y = (X W^T + b)*oscale
__global__ __launch_bounds__(256)
void proj16_kernel(const unsigned short* __restrict__ X,
                   const unsigned short* __restrict__ W,
                   const float* __restrict__ bias,
                   unsigned short* __restrict__ Ybf,
                   float* __restrict__ Yf32,
                   int transpose, float oscale) {
    const int lane = threadIdx.x & 31;
    const int wave = threadIdx.x >> 5;
    const int tile = blockIdx.x * 8 + wave;
    const int rowbase = (tile >> 4) * 16;
    const int chbase  = (tile & 15) * 16;
    const int l = lane & 15, h = (lane >> 4) & 1;

    v8f acc = {0.f, 0.f, 0.f, 0.f, 0.f, 0.f, 0.f, 0.f};
    #pragma unroll
    for (int cc = 0; cc < 8; ++cc) {
        v16bf a = load_frag(X + (size_t)rowbase * NC + cc * 32, NC, lane);
        v16bf b = load_frag(W + (size_t)chbase  * NC + cc * 32, NC, lane);
        acc = __builtin_amdgcn_wmma_f32_16x16x32_bf16(false, a, false, b,
                                                      (short)0, acc, false, false);
    }
    const float bn = bias[chbase + l];
    if (Yf32) {
        #pragma unroll
        for (int r = 0; r < 8; ++r)
            Yf32[(size_t)(rowbase + r + 8 * h) * NC + chbase + l] = (acc[r] + bn) * oscale;
    } else if (transpose) {
        #pragma unroll
        for (int r = 0; r < 8; ++r)
            Ybf[(size_t)(chbase + l) * NM + rowbase + 8 * h + r] = f2bf((acc[r] + bn) * oscale);
    } else {
        #pragma unroll
        for (int r = 0; r < 8; ++r)
            Ybf[(size_t)(rowbase + r + 8 * h) * NC + chbase + l] = f2bf((acc[r] + bn) * oscale);
    }
}

// -------------------------------- cooperative async K/V staging (global->LDS)
__device__ __forceinline__ void stage_tiles_async(const unsigned short* __restrict__ Kb,
                                                  const unsigned short* __restrict__ Vt,
                                                  unsigned int ldsK, unsigned int ldsV,
                                                  int kb, int tid) {
    // K block: 32 keys x 256 ch = contiguous 16KB; 256 threads x 4 x b128.
    const unsigned short* gK = Kb + (size_t)kb * NC;
    #pragma unroll
    for (int i = 0; i < 4; ++i) {
        const unsigned int off = (unsigned int)((i * 256 + tid) * 16);
        async_ld_b128(ldsK + off, gK, off);
    }
    // Vt block: 256 rows x 64B (stride NM*2B); one row per thread, 4 x b128.
    #pragma unroll
    for (int i = 0; i < 4; ++i) {
        const unsigned int goff = (unsigned int)(((size_t)tid * NM + kb) * 2 + i * 16);
        async_ld_b128(ldsV + (unsigned int)(tid * 64 + i * 16), Vt, goff);
    }
}

// -------------------------------------------------- fused epipolar flash attn
// grid.x: 64 q-groups (8 waves x 16 rows); grid.y: 4 key splits.
// Partial (unnormalized) O in bf16 + per-row {max, sum, disp-num, any} stats.
__global__ __launch_bounds__(256)
void epi_attn_kernel(const unsigned short* __restrict__ Qb,
                     const unsigned short* __restrict__ Kb,
                     const unsigned short* __restrict__ Vt,
                     const float* __restrict__ kptsL,
                     const float* __restrict__ kptsR,
                     unsigned short* __restrict__ Opart,
                     float* __restrict__ Ms, float* __restrict__ Ls,
                     float* __restrict__ Ds, float* __restrict__ As) {
    __shared__ unsigned short Ksh[2][32][NC];   // 2 x 16KB (double-buffered)
    __shared__ unsigned short Vsh[2][NC][32];   // 2 x 16KB (transposed V tile)
    __shared__ unsigned short pbuf[8][16][32];  // 8KB per-wave P staging

    const int tid = threadIdx.x;
    const int lane = tid & 31, wave = tid >> 5;
    const int l = lane & 15, h = (lane >> 4) & 1;
    const int split = blockIdx.y;
    const int kb0 = split * KSPAN;
    const int qbase = (blockIdx.x * 8 + wave) * 16;

    const unsigned int ldsK[2] = { (unsigned int)(uintptr_t)&Ksh[0][0][0],
                                   (unsigned int)(uintptr_t)&Ksh[1][0][0] };
    const unsigned int ldsV[2] = { (unsigned int)(uintptr_t)&Vsh[0][0][0],
                                   (unsigned int)(uintptr_t)&Vsh[1][0][0] };

    v16bf qf[8];
    #pragma unroll
    for (int cc = 0; cc < 8; ++cc)
        qf[cc] = load_frag(Qb + (size_t)qbase * NC + cc * 32, NC, lane);

    float uL[8], vL[8];
    #pragma unroll
    for (int r = 0; r < 8; ++r) {
        int row = qbase + r + 8 * h;
        uL[r] = kptsL[2 * row];
        vL[r] = kptsL[2 * row + 1];
    }

    v8f acc[16];
    #pragma unroll
    for (int t = 0; t < 16; ++t)
        acc[t] = (v8f){0.f, 0.f, 0.f, 0.f, 0.f, 0.f, 0.f, 0.f};
    float mrow[8], lrow[8], drow[8];
    int anyr[8];
    #pragma unroll
    for (int r = 0; r < 8; ++r) { mrow[r] = -3.0e38f; lrow[r] = 0.f; drow[r] = 0.f; anyr[r] = 0; }

    stage_tiles_async(Kb, Vt, ldsK[0], ldsV[0], kb0, tid);
    wait_async0();
    __syncthreads();

    const int NIT = KSPAN / 32;   // 64
    for (int it = 0; it < NIT; ++it) {
        const int kb = kb0 + it * 32;
        const int cur = it & 1, nxt = cur ^ 1;
        if (it + 1 < NIT)
            stage_tiles_async(Kb, Vt, ldsK[nxt], ldsV[nxt], kb + 32, tid);

        // S = Q K^T for 32 keys (two 16x16 tiles) from LDS
        v8f s0 = {0.f, 0.f, 0.f, 0.f, 0.f, 0.f, 0.f, 0.f};
        v8f s1 = {0.f, 0.f, 0.f, 0.f, 0.f, 0.f, 0.f, 0.f};
        #pragma unroll
        for (int cc = 0; cc < 8; ++cc) {
            v16bf k0 = load_frag(&Ksh[cur][0][0]  + cc * 32, NC, lane);
            v16bf k1 = load_frag(&Ksh[cur][16][0] + cc * 32, NC, lane);
            s0 = __builtin_amdgcn_wmma_f32_16x16x32_bf16(false, qf[cc], false, k0,
                                                         (short)0, s0, false, false);
            s1 = __builtin_amdgcn_wmma_f32_16x16x32_bf16(false, qf[cc], false, k1,
                                                         (short)0, s1, false, false);
        }
        const float uR0 = kptsR[2 * (kb + l)],      vR0 = kptsR[2 * (kb + l) + 1];
        const float uR1 = kptsR[2 * (kb + 16 + l)], vR1 = kptsR[2 * (kb + 16 + l) + 1];

        float p0[8], p1[8], alpha[8];
        #pragma unroll
        for (int r = 0; r < 8; ++r) {
            const float du0 = uL[r] - uR0, dv0 = fabsf(vL[r] - vR0);
            const float du1 = uL[r] - uR1, dv1 = fabsf(vL[r] - vR1);
            const bool m0 = (dv0 < VTOL) && (du0 > 0.f) && (du0 < MAXD);
            const bool m1 = (dv1 < VTOL) && (du1 > 0.f) && (du1 < MAXD);
            anyr[r] |= (int)m0 | (int)m1;
            const float e0 = m0 ? s0[r] : NEGV;   // scale pre-folded into Q
            const float e1 = m1 ? s1[r] : NEGV;
            const float mx = red_max16(fmaxf(e0, e1));
            const float nm = fmaxf(mrow[r], mx);
            alpha[r] = __expf(mrow[r] - nm);
            mrow[r] = nm;
            p0[r] = __expf(e0 - nm);
            p1[r] = __expf(e1 - nm);
            const float rs = red_sum16(p0[r] + p1[r]);
            const float ds = red_sum16(p0[r] * du0 + p1[r] * du1);
            lrow[r] = lrow[r] * alpha[r] + rs;
            drow[r] = drow[r] * alpha[r] + ds;
        }
        #pragma unroll
        for (int t = 0; t < 16; ++t)
            #pragma unroll
            for (int r = 0; r < 8; ++r) acc[t][r] *= alpha[r];

        // Stage P (C/D layout -> row-major bf16 LDS) and reload as A-fragment
        #pragma unroll
        for (int r = 0; r < 8; ++r) {
            pbuf[wave][r + 8 * h][l]      = f2bf(p0[r]);
            pbuf[wave][r + 8 * h][16 + l] = f2bf(p1[r]);
        }
        asm volatile("s_wait_dscnt 0" ::: "memory");   // same-wave LDS RAW
        v16bf pf = load_frag(&pbuf[wave][0][0], 32, lane);

        // O += P · V from LDS (Vsh holds transposed V tile)
        #pragma unroll
        for (int t = 0; t < 16; ++t) {
            v16bf vb = load_frag(&Vsh[cur][t * 16][0], 32, lane);
            acc[t] = __builtin_amdgcn_wmma_f32_16x16x32_bf16(false, pf, false, vb,
                                                             (short)0, acc[t], false, false);
        }
        wait_async0();     // nxt buffer fully landed (had full compute to hide)
        __syncthreads();   // cur reads done before it gets re-staged
    }

    // Epilogue: unnormalized partials + per-row stats for this split
    #pragma unroll
    for (int r = 0; r < 8; ++r) {
        const int row = qbase + r + 8 * h;
        #pragma unroll
        for (int t = 0; t < 16; ++t)
            Opart[((size_t)split * NQ + row) * NC + t * 16 + l] = f2bf(acc[t][r]);
        int a = anyr[r];
        #pragma unroll
        for (int o = 1; o < 16; o <<= 1) a |= __shfl_xor(a, o, 32);
        if (l == 0) {
            Ms[split * NQ + row] = mrow[r];
            Ls[split * NQ + row] = lrow[r];
            Ds[split * NQ + row] = drow[r];
            As[split * NQ + row] = a ? 1.0f : 0.0f;
        }
    }
}

// ------------------------------------------------------ split-K merge kernel
__global__ __launch_bounds__(256)
void merge_kernel(const unsigned short* __restrict__ Opart,
                  const float* __restrict__ Ms, const float* __restrict__ Ls,
                  const float* __restrict__ Ds, const float* __restrict__ As,
                  unsigned short* __restrict__ Mout,
                  float* __restrict__ dout, float* __restrict__ cout) {
    const int row = blockIdx.x;
    const int ch = threadIdx.x;
    float M = -3.0e38f;
    #pragma unroll
    for (int s = 0; s < NSPLIT; ++s) M = fmaxf(M, Ms[s * NQ + row]);
    float w[NSPLIT], lsum = 0.f, dsum = 0.f, any = 0.f;
    #pragma unroll
    for (int s = 0; s < NSPLIT; ++s) {
        w[s] = __expf(Ms[s * NQ + row] - M);
        lsum += w[s] * Ls[s * NQ + row];
        dsum += w[s] * Ds[s * NQ + row];
        any = fmaxf(any, As[s * NQ + row]);
    }
    const float inv = 1.0f / lsum;
    float o = 0.f;
    #pragma unroll
    for (int s = 0; s < NSPLIT; ++s)
        o += w[s] * bf2f(Opart[((size_t)s * NQ + row) * NC + ch]);
    Mout[(size_t)row * NC + ch] = f2bf(o * inv);
    if (ch == 0) {
        dout[row] = dsum * inv;
        cout[row] = any;
    }
}

// ------------------------------------------------------------------ launcher
extern "C" void kernel_launch(void* const* d_in, const int* in_sizes, int n_in,
                              void* d_out, int out_size, void* d_ws, size_t ws_size,
                              hipStream_t stream) {
    const float* nodes_L = (const float*)d_in[0];
    const float* nodes_R = (const float*)d_in[1];
    const float* kpts_L  = (const float*)d_in[2];
    const float* kpts_R  = (const float*)d_in[3];
    const float* Wq = (const float*)d_in[4];  const float* bq = (const float*)d_in[5];
    const float* Wk = (const float*)d_in[6];  const float* bk = (const float*)d_in[7];
    const float* Wv = (const float*)d_in[8];  const float* bv = (const float*)d_in[9];
    const float* Wm = (const float*)d_in[10]; const float* bm = (const float*)d_in[11];

    float* out     = (float*)d_out;               // [N, C]
    float* dispOut = out + (size_t)NQ * NC;       // [N]
    float* confOut = dispOut + NQ;                // [N]

    // Workspace layout
    unsigned short* ws = (unsigned short*)d_ws;
    const size_t NCel = (size_t)NQ * NC;          // 2,097,152 elements
    const size_t WEL  = (size_t)NC * NC;          // 65,536 elements
    unsigned short* nLb = ws;
    unsigned short* nRb = nLb + NCel;
    unsigned short* Wqb = nRb + NCel;
    unsigned short* Wkb = Wqb + WEL;
    unsigned short* Wvb = Wkb + WEL;
    unsigned short* Wmb = Wvb + WEL;
    unsigned short* Qb  = Wmb + WEL;
    unsigned short* Kb  = Qb + NCel;
    unsigned short* Vtb = Kb + NCel;
    unsigned short* Mb  = Vtb + NCel;
    unsigned short* Opart = Mb + NCel;            // [NSPLIT][N][C] bf16
    float* stats = (float*)(Opart + (size_t)NSPLIT * NCel);
    float* Ms = stats;                            // [NSPLIT][N]
    float* Ls = Ms + (size_t)NSPLIT * NQ;
    float* Ds = Ls + (size_t)NSPLIT * NQ;
    float* As = Ds + (size_t)NSPLIT * NQ;

    // 1) f32 -> bf16 conversions
    {
        int n = (int)NCel;
        cvt_bf16_kernel<<<(n + 255) / 256, 256, 0, stream>>>(nodes_L, nLb, n);
        cvt_bf16_kernel<<<(n + 255) / 256, 256, 0, stream>>>(nodes_R, nRb, n);
        int w = (int)WEL;
        cvt_bf16_kernel<<<(w + 255) / 256, 256, 0, stream>>>(Wq, Wqb, w);
        cvt_bf16_kernel<<<(w + 255) / 256, 256, 0, stream>>>(Wk, Wkb, w);
        cvt_bf16_kernel<<<(w + 255) / 256, 256, 0, stream>>>(Wv, Wvb, w);
        cvt_bf16_kernel<<<(w + 255) / 256, 256, 0, stream>>>(Wm, Wmb, w);
    }

    // 2) projections: Q (pre-scaled by 1/sqrt(C)), K row-major, V transposed
    const int projBlocks = (NQ / 16) * (NC / 16) / 8;   // 1024
    proj16_kernel<<<projBlocks, 256, 0, stream>>>(nLb, Wqb, bq, Qb,  nullptr, 0, SCALE);
    proj16_kernel<<<projBlocks, 256, 0, stream>>>(nRb, Wkb, bk, Kb,  nullptr, 0, 1.0f);
    proj16_kernel<<<projBlocks, 256, 0, stream>>>(nRb, Wvb, bv, Vtb, nullptr, 1, 1.0f);

    // 3) fused flash attention, split-K over 4 key ranges (256 workgroups)
    dim3 agrid(NQ / 16 / 8, NSPLIT, 1);
    epi_attn_kernel<<<agrid, 256, 0, stream>>>(Qb, Kb, Vtb, kpts_L, kpts_R,
                                               Opart, Ms, Ls, Ds, As);

    // 4) merge partials -> matched bf16 + disparity + confidence
    merge_kernel<<<NQ, 256, 0, stream>>>(Opart, Ms, Ls, Ds, As, Mb, dispOut, confOut);

    // 5) output projection -> f32 d_out
    proj16_kernel<<<projBlocks, 256, 0, stream>>>(Mb, Wmb, bm, nullptr, out, 0, 1.0f);
}
